// AttentiveItemToVec_32873679683782
// MI455X (gfx1250) — matrix operand order repaired and checked by reading, not compiled, not run on hardware
//
#include <hip/hip_runtime.h>
#include <hip/hip_bf16.h>

// ---------------------------------------------------------------------------
// AttentiveItemToVec, fully fused per batch element on gfx1250 (MI455X).
// All six GEMMs run as v_wmma_f32_16x16x32_bf16 tiles; all intermediates
// ([J,D],[M,D],[J,M],[M,E],[J,E]) live in 160KB of LDS (2 WGs / 320KB WGP).
// ---------------------------------------------------------------------------

typedef __attribute__((ext_vector_type(16))) __bf16 v16bf;
typedef __attribute__((ext_vector_type(8)))  float  v8f;

#define B_  512
#define J_  101
#define M_  50
#define E_  128
#define D_  60
#define P_  5120
#define JP  112   // J padded to 7*16
#define MP  64    // M padded to 4*16
#define DP  64    // D padded to 4*16

// ---- LDS layout (byte offsets), phase-overlaid ----------------------------
#define OFF_U     0        // u_bf   [64][128]  bf16  16384      (stages 1-6)
#define OFF_V     16384    // v_bf   [112][128] bf16  28672      (1-3)
                           //   cos_f[112][64]  f32   28672      (4-5)
#define OFF_W     45056    // At_bf  [64][128]  bf16  16384      (1-3)
#define OFF_CPF   61440    //   cp_f [64][64]   f32   16384      (2)
                           // W2 (Bc then R) [128][128] bf16 32768 (6 / 8)
#define OFF_AC    77824    // Ac_bf  [64][128]  bf16  16384      (1-2)
                           //   attn_bf[112][64] bf16 14336      (5-7)
#define OFF_TPF   94208    // tp_f   [112][64]  f32   28672      (3)
                           //   alpha_bf[112][128] bf16 28672    (7-8)
#define OFF_CPBF  122880   // cp_bf  [64][64]   bf16  8192       (2-4)
#define OFF_TPBF  131072   // tp_bf  [112][64]  bf16  14336      (3-4)
#define OFF_BUT   145408   // buT_bf [128][64]  bf16  16384      (6-7)
#define OFF_NC    161792   // nc[64] f32
#define OFF_NT    162048   // nt[112] f32
#define OFF_BIAS  162560   // bias/mask[64] f32
#define OFF_RB    162816   // R_b[128] f32
#define SMEM_BYTES 163840  // 160 KB

__device__ __forceinline__ __bf16 f2bf(float f) {
    unsigned u = __builtin_bit_cast(unsigned, f);
    unsigned short s = (unsigned short)((u + 0x7fffu + ((u >> 16) & 1u)) >> 16);
    return __builtin_bit_cast(__bf16, s);
}

__device__ __forceinline__ v8f wmma_bf16(v16bf a, v16bf b, v8f c) {
    return __builtin_amdgcn_wmma_f32_16x16x32_bf16(
        false, a, false, b, (short)0, c, false, false);
}

// A fragment: 16x32 bf16 tile (row0,k0) of row-major [rows][ld] matrix.
// lane<16: row=lane,   K k0+0..15
// lane>=16: row=lane-16, K k0+16..31
__device__ __forceinline__ v16bf load_a(const __bf16* base, int row0, int k0,
                                        int ld, int lane) {
    int r = lane & 15, h = lane >> 4;
    const __bf16* p = base + (row0 + r) * ld + k0 + h * 16;
    v16bf a;
#pragma unroll
    for (int i = 0; i < 8; ++i) a[i] = p[i];
#pragma unroll
    for (int i = 0; i < 8; ++i) a[8 + i] = p[8 + i];
    return a;
}

// B fragment: 32x16 tile of B, supplied as row-major B^T [N][ld] (=weights).
// lane<16: col=lane, K k0+0..15 ; lane>=16: col=lane-16, K k0+16..31
__device__ __forceinline__ v16bf load_b(const __bf16* baseT, int n0, int k0,
                                        int ld, int lane) {
    int n = lane & 15, h = lane >> 4;
    const __bf16* p = baseT + (n0 + n) * ld + k0 + h * 16;
    v16bf b;
#pragma unroll
    for (int i = 0; i < 16; ++i) b[i] = p[i];
    return b;
}

__global__ __launch_bounds__(256) void
aiv_fused_kernel(const int* __restrict__ titems, const int* __restrict__ citems,
                 const int* __restrict__ padids,
                 const float* __restrict__ t_emb, const float* __restrict__ c_emb,
                 const float* __restrict__ Ac_w, const float* __restrict__ Ac_b,
                 const float* __restrict__ At_w, const float* __restrict__ At_b,
                 const float* __restrict__ Bc_w, const float* __restrict__ Bc_b,
                 const float* __restrict__ R_w,  const float* __restrict__ R_b,
                 float* __restrict__ out) {
    extern __shared__ char smem[];
    __bf16* u_bf    = (__bf16*)(smem + OFF_U);
    __bf16* v_bf    = (__bf16*)(smem + OFF_V);
    float*  cos_f   = (float*)(smem + OFF_V);
    __bf16* At_bf   = (__bf16*)(smem + OFF_W);
    float*  cp_f    = (float*)(smem + OFF_CPF);
    __bf16* W2      = (__bf16*)(smem + OFF_W);
    __bf16* Ac_bf   = (__bf16*)(smem + OFF_AC);
    __bf16* attn_bf = (__bf16*)(smem + OFF_AC);
    float*  tp_f    = (float*)(smem + OFF_TPF);
    __bf16* alpha_bf= (__bf16*)(smem + OFF_TPF);
    __bf16* cp_bf   = (__bf16*)(smem + OFF_CPBF);
    __bf16* tp_bf   = (__bf16*)(smem + OFF_TPBF);
    __bf16* buT_bf  = (__bf16*)(smem + OFF_BUT);
    float*  nc      = (float*)(smem + OFF_NC);
    float*  nt      = (float*)(smem + OFF_NT);
    float*  bias    = (float*)(smem + OFF_BIAS);
    float*  Rb_s    = (float*)(smem + OFF_RB);

    const int tid  = threadIdx.x;
    const int lane = tid & 31;
    const int wv   = tid >> 5;
    const int b    = blockIdx.x;
    const float NEG_INF = -__builtin_inff();
    const float EPS = 1e-6f;

    // ---- Stage 1: gather v,u -> bf16 LDS; stage At,Ac weights; init bias --
    const float4* t4  = (const float4*)t_emb;
    const float4* c4  = (const float4*)c_emb;
    const float4* at4 = (const float4*)At_w;
    const float4* ac4 = (const float4*)Ac_w;
    for (int p = tid; p < JP * (E_ / 4); p += 256) {     // v: 112x128
        int j = p >> 5, q = p & 31;
        __bf16* dst = v_bf + j * E_ + q * 4;
        if (j < J_) {
            int row = titems[b * J_ + j];
            float4 x = t4[(size_t)row * 32 + q];
            dst[0] = f2bf(x.x); dst[1] = f2bf(x.y);
            dst[2] = f2bf(x.z); dst[3] = f2bf(x.w);
        } else {
            dst[0] = dst[1] = dst[2] = dst[3] = f2bf(0.f);
        }
    }
    for (int p = tid; p < MP * (E_ / 4); p += 256) {     // u: 64x128
        int m = p >> 5, q = p & 31;
        __bf16* dst = u_bf + m * E_ + q * 4;
        if (m < M_) {
            int row = citems[b * M_ + m];
            float4 x = c4[(size_t)row * 32 + q];
            dst[0] = f2bf(x.x); dst[1] = f2bf(x.y);
            dst[2] = f2bf(x.z); dst[3] = f2bf(x.w);
        } else {
            dst[0] = dst[1] = dst[2] = dst[3] = f2bf(0.f);
        }
    }
    for (int p = tid; p < DP * (E_ / 4); p += 256) {     // At, Ac: 64x128
        int r = p >> 5, q = p & 31;
        __bf16* d1 = At_bf + r * E_ + q * 4;
        __bf16* d2 = Ac_bf + r * E_ + q * 4;
        if (r < D_) {
            float4 x = at4[r * 32 + q];
            d1[0] = f2bf(x.x); d1[1] = f2bf(x.y); d1[2] = f2bf(x.z); d1[3] = f2bf(x.w);
            float4 y = ac4[r * 32 + q];
            d2[0] = f2bf(y.x); d2[1] = f2bf(y.y); d2[2] = f2bf(y.z); d2[3] = f2bf(y.w);
        } else {
            d1[0]=d1[1]=d1[2]=d1[3]=f2bf(0.f);
            d2[0]=d2[1]=d2[2]=d2[3]=f2bf(0.f);
        }
    }
    if (tid < MP) bias[tid] = (tid < M_) ? 0.f : NEG_INF;
    __syncthreads();

    // ---- Stage 2: cp = u @ Ac^T  (4x4 tiles, K=128) + pad-mask scan -------
    for (int t = wv; t < 16; t += 8) {
        int mt = t & 3, ntl = t >> 2;
        v8f acc = {};
#pragma unroll
        for (int kt = 0; kt < 4; ++kt) {
            v16bf a = load_a(u_bf,  mt * 16, kt * 32, E_, lane);
            v16bf bb = load_b(Ac_bf, ntl * 16, kt * 32, E_, lane);
            acc = wmma_bf16(a, bb, acc);
        }
        int col = lane & 15, rb0 = (lane >> 4) * 8;
#pragma unroll
        for (int r = 0; r < 8; ++r)
            cp_f[(mt * 16 + rb0 + r) * DP + ntl * 16 + col] = acc[r];
    }
    for (int i = tid; i < P_; i += 256)                  // mask (b,pos) pairs
        if (padids[i] == b) bias[padids[P_ + i]] = NEG_INF;
    __syncthreads();

    // ---- Stage 2b: cp += Ac_b, cp->bf16, row norms nc ---------------------
    for (int idx = tid; idx < MP * DP; idx += 256) {
        int d = idx & (DP - 1);
        float x = cp_f[idx] + ((d < D_) ? Ac_b[d] : 0.f);
        cp_f[idx] = x;
        cp_bf[idx] = f2bf(x);
    }
    __syncthreads();
    if (tid < MP) {
        float s = 0.f;
        for (int d = 0; d < DP; ++d) { float x = cp_f[tid * DP + d]; s += x * x; }
        nc[tid] = fmaxf(sqrtf(s), EPS);
    }
    __syncthreads();

    // ---- Stage 3: tp = v @ At^T  (7x4 tiles, K=128) -----------------------
    for (int t = wv; t < 28; t += 8) {
        int mt = t % 7, ntl = t / 7;
        v8f acc = {};
#pragma unroll
        for (int kt = 0; kt < 4; ++kt) {
            v16bf a = load_a(v_bf,  mt * 16, kt * 32, E_, lane);
            v16bf bb = load_b(At_bf, ntl * 16, kt * 32, E_, lane);
            acc = wmma_bf16(a, bb, acc);
        }
        int col = lane & 15, rb0 = (lane >> 4) * 8;
#pragma unroll
        for (int r = 0; r < 8; ++r)
            tp_f[(mt * 16 + rb0 + r) * DP + ntl * 16 + col] = acc[r];
    }
    __syncthreads();
    for (int idx = tid; idx < JP * DP; idx += 256) {
        int d = idx & (DP - 1);
        float x = tp_f[idx] + ((d < D_) ? At_b[d] : 0.f);
        tp_f[idx] = x;
        tp_bf[idx] = f2bf(x);
    }
    __syncthreads();
    if (tid < JP) {
        float s = 0.f;
        for (int d = 0; d < DP; ++d) { float x = tp_f[tid * DP + d]; s += x * x; }
        nt[tid] = fmaxf(sqrtf(s), EPS);
    }
    __syncthreads();

    // ---- Stage 4: dot = tp @ cp^T (7x4 tiles, K=64) -> cos_f (over v) -----
    for (int t = wv; t < 28; t += 8) {
        int mt = t % 7, ntl = t / 7;
        v8f acc = {};
#pragma unroll
        for (int kt = 0; kt < 2; ++kt) {
            v16bf a = load_a(tp_bf, mt * 16, kt * 32, DP, lane);
            v16bf bb = load_b(cp_bf, ntl * 16, kt * 32, DP, lane);
            acc = wmma_bf16(a, bb, acc);
        }
        int col = lane & 15, rb0 = (lane >> 4) * 8;
#pragma unroll
        for (int r = 0; r < 8; ++r)
            cos_f[(mt * 16 + rb0 + r) * MP + ntl * 16 + col] = acc[r];
    }
    __syncthreads();

    // ---- Stage 5: masked softmax over M (wave per row) -> attn_bf ---------
    for (int j = wv; j < J_; j += 8) {
        float ntj = nt[j];
        float v0 = cos_f[j * MP + lane]      / (ntj * nc[lane])      + bias[lane];
        float v1 = cos_f[j * MP + lane + 32] / (ntj * nc[lane + 32]) + bias[lane + 32];
        float mx = fmaxf(v0, v1);
#pragma unroll
        for (int o = 16; o; o >>= 1) mx = fmaxf(mx, __shfl_xor(mx, o, 32));
        float e0 = __expf(v0 - mx), e1 = __expf(v1 - mx);
        float s = e0 + e1;
#pragma unroll
        for (int o = 16; o; o >>= 1) s += __shfl_xor(s, o, 32);
        float inv = 1.f / s;
        attn_bf[j * MP + lane]      = f2bf(e0 * inv);
        attn_bf[j * MP + lane + 32] = f2bf(e1 * inv);
    }
    for (int j = J_ + wv; j < JP; j += 8) {              // zero padded rows
        attn_bf[j * MP + lane]      = f2bf(0.f);
        attn_bf[j * MP + lane + 32] = f2bf(0.f);
    }
    __syncthreads();

    // ---- Stage 6: W2 <- Bc (bf16); buT = (u @ Bc^T + Bc_b)^T --------------
    for (int idx = tid; idx < E_ * E_; idx += 256) W2[idx] = f2bf(Bc_w[idx]);
    if (tid < E_) Rb_s[tid] = R_b[tid];
    __syncthreads();
    for (int t = wv; t < 32; t += 8) {                   // 4 x 8 tiles, K=128
        int mt = t & 3, ntl = t >> 2;
        v8f acc = {};
#pragma unroll
        for (int kt = 0; kt < 4; ++kt) {
            v16bf a = load_a(u_bf, mt * 16, kt * 32, E_, lane);
            v16bf bb = load_b(W2,  ntl * 16, kt * 32, E_, lane);
            acc = wmma_bf16(a, bb, acc);
        }
        int col = lane & 15, rb0 = (lane >> 4) * 8;
        float bb = Bc_b[ntl * 16 + col];
#pragma unroll
        for (int r = 0; r < 8; ++r) {                    // transposed store
            int m = mt * 16 + rb0 + r, e = ntl * 16 + col;
            buT_bf[e * MP + m] = f2bf(acc[r] + bb);
        }
    }
    __syncthreads();

    // ---- Stage 7: alpha = attn @ bu (7x8 tiles, K=64); W2 <- R ------------
    for (int t = wv; t < 56; t += 8) {
        int mt = t % 7, ntl = t / 7;
        v8f acc = {};
#pragma unroll
        for (int kt = 0; kt < 2; ++kt) {
            v16bf a = load_a(attn_bf, mt * 16, kt * 32, MP, lane);
            v16bf bb = load_b(buT_bf, ntl * 16, kt * 32, MP, lane);
            acc = wmma_bf16(a, bb, acc);
        }
        int col = lane & 15, rb0 = (lane >> 4) * 8;
#pragma unroll
        for (int r = 0; r < 8; ++r)
            alpha_bf[(mt * 16 + rb0 + r) * E_ + ntl * 16 + col] = f2bf(acc[r]);
    }
    for (int idx = tid; idx < E_ * E_; idx += 256) W2[idx] = f2bf(R_w[idx]);
    __syncthreads();

    // ---- Stage 8: out = alpha @ R^T + R_b (7x8 tiles, K=128) -> HBM -------
    for (int t = wv; t < 56; t += 8) {
        int mt = t % 7, ntl = t / 7;
        v8f acc = {};
#pragma unroll
        for (int kt = 0; kt < 4; ++kt) {
            v16bf a = load_a(alpha_bf, mt * 16, kt * 32, E_, lane);
            v16bf bb = load_b(W2,      ntl * 16, kt * 32, E_, lane);
            acc = wmma_bf16(a, bb, acc);
        }
        int col = lane & 15, rb0 = (lane >> 4) * 8;
        float rb = Rb_s[ntl * 16 + col];
#pragma unroll
        for (int r = 0; r < 8; ++r) {
            int j = mt * 16 + rb0 + r;
            if (j < J_)
                out[((size_t)b * J_ + j) * E_ + ntl * 16 + col] = acc[r] + rb;
        }
    }
}

extern "C" void kernel_launch(void* const* d_in, const int* in_sizes, int n_in,
                              void* d_out, int out_size, void* d_ws, size_t ws_size,
                              hipStream_t stream) {
    (void)in_sizes; (void)n_in; (void)out_size; (void)d_ws; (void)ws_size;
    const int*   titems = (const int*)d_in[0];
    const int*   citems = (const int*)d_in[1];
    const int*   padids = (const int*)d_in[2];
    const float* t_emb  = (const float*)d_in[3];
    const float* c_emb  = (const float*)d_in[4];
    const float* Ac_w   = (const float*)d_in[5];
    const float* Ac_b   = (const float*)d_in[6];
    const float* At_w   = (const float*)d_in[7];
    const float* At_b   = (const float*)d_in[8];
    const float* Bc_w   = (const float*)d_in[9];
    const float* Bc_b   = (const float*)d_in[10];
    const float* R_w    = (const float*)d_in[11];
    const float* R_b    = (const float*)d_in[12];
    float* out = (float*)d_out;

    aiv_fused_kernel<<<dim3(B_), dim3(256), SMEM_BYTES, stream>>>(
        titems, citems, padids, t_emb, c_emb,
        Ac_w, Ac_b, At_b ? At_w : At_w, At_b, Bc_w, Bc_b, R_w, R_b, out);
}
